// multiheadAttentionNLP_8211977470028
// MI455X (gfx1250) — matrix operand order repaired and checked by reading, not compiled, not run on hardware
//
#include <hip/hip_runtime.h>
#include <hip/hip_bf16.h>

typedef __attribute__((ext_vector_type(16))) __bf16 v16bf;
typedef __attribute__((ext_vector_type(8)))  float  v8f;
typedef __attribute__((ext_vector_type(4)))  unsigned int t_v4u;
typedef __attribute__((ext_vector_type(8)))  int          t_v8i;
typedef __attribute__((ext_vector_type(4)))  int          t_v4i;

#define WMMA_BF16(a, b, c) \
  __builtin_amdgcn_wmma_f32_16x16x32_bf16(false, (a), false, (b), (short)0, (c), false, false)

union BV16 { v16bf v; uint4 q[2]; __bf16 e[16]; };
union PK8  { __bf16 e[8]; uint4 q; };

// ---------------------------------------------------------------------------
// TDM: 2-D bf16 tile load from global -> LDS (one descriptor, count=1).
// Optional LDS row padding (pad_interval/pad_amount codes per ISA D# group1).
// ---------------------------------------------------------------------------
__device__ __forceinline__ void tdm_load_2d_bf16(unsigned lds_addr, const void* gaddr,
                                                 unsigned tensor_d0, unsigned tensor_d1,
                                                 unsigned stride0,
                                                 unsigned tile_d0, unsigned tile_d1,
                                                 unsigned pad_interval_code,
                                                 unsigned pad_amount_code,
                                                 bool pad_en) {
  unsigned long long ga = (unsigned long long)gaddr;
  t_v4u g0;
  g0[0] = 1u;                                            // count=1, user descriptor
  g0[1] = lds_addr;                                      // LDS byte address
  g0[2] = (unsigned)ga;                                  // global_addr[31:0]
  g0[3] = (unsigned)((ga >> 32) & 0x01FFFFFFu) | 0x80000000u;  // addr[56:32] | type=2
  t_v8i g1;
  unsigned w0 = (1u << 16);                              // data_size = 2 bytes
  if (pad_en) w0 |= (1u << 20) | (pad_interval_code << 22) | (pad_amount_code << 25);
  g1[0] = (int)w0;
  g1[1] = (int)((tensor_d0 & 0xFFFFu) << 16);            // tensor_dim0[15:0]
  g1[2] = (int)((tensor_d0 >> 16) | ((tensor_d1 & 0xFFFFu) << 16));
  g1[3] = (int)((tensor_d1 >> 16) | (tile_d0 << 16));    // tile_dim0
  g1[4] = (int)(tile_d1 & 0xFFFFu);                      // tile_dim1 (tile_dim2=0)
  g1[5] = (int)stride0;                                  // tensor_dim0_stride[31:0]
  g1[6] = 0;
  g1[7] = 0;
  t_v4i z4 = {0, 0, 0, 0};
#if defined(__clang_major__) && (__clang_major__ >= 23)
  t_v8i z8 = {0, 0, 0, 0, 0, 0, 0, 0};
  __builtin_amdgcn_tensor_load_to_lds(g0, g1, z4, z4, z8, 0);
#else
  __builtin_amdgcn_tensor_load_to_lds(g0, g1, z4, z4, 0);
#endif
}

// ---------------------------------------------------------------------------
// Prep: fp32 -> bf16 elementwise (nodes)
// ---------------------------------------------------------------------------
__global__ void k_cvt(const float* __restrict__ x, __bf16* __restrict__ y, int n) {
  int i = (blockIdx.x * 256 + threadIdx.x) * 4;
  if (i + 3 < n) {
    float4 v = *(const float4*)(x + i);
    y[i + 0] = (__bf16)v.x;
    y[i + 1] = (__bf16)v.y;
    y[i + 2] = (__bf16)v.z;
    y[i + 3] = (__bf16)v.w;
  }
}

// Prep: transpose+convert a 512x512 fp32 weight -> bf16 W^T
__global__ void k_tw(const float* __restrict__ w, __bf16* __restrict__ wt) {
  int i = blockIdx.x * 256 + threadIdx.x;
  int k = i >> 9;
  int n = i & 511;
  wt[n * 512 + k] = (__bf16)w[i];
}

// ---------------------------------------------------------------------------
// GEMM, transposed orientation (see round-1 analysis). mode 0: bf16 [B,H,S,64],
// mode 1: fp32 row-major [M,512].
// ---------------------------------------------------------------------------
__global__ void k_gemmT(const __bf16* __restrict__ Wt, const __bf16* __restrict__ X,
                        const float* __restrict__ bias, void* __restrict__ outp, int mode) {
  const int wave = threadIdx.x >> 5;
  const int lane = threadIdx.x & 31;
  const int c    = lane & 15;
  const int hi   = (lane >> 4) & 1;
  const int task = blockIdx.x * 8 + wave;
  const int m0   = (task >> 3) << 4;
  const int n0   = (task & 7)  << 6;

  v8f acc[4] = {};
  const __bf16* xrow = X + (size_t)(m0 + c) * 512 + hi * 16;
  for (int ks = 0; ks < 512; ks += 32) {
    BV16 b;
    b.v = *(const v16bf*)(xrow + ks);
#pragma unroll
    for (int nt = 0; nt < 4; ++nt) {
      const __bf16* arow = Wt + (size_t)(n0 + nt * 16 + c) * 512 + ks + hi * 16;
      BV16 a;
      a.q[0] = *(const uint4*)(arow);
      a.q[1] = *(const uint4*)(arow + 8);
      acc[nt] = WMMA_BF16(a.v, b.v, acc[nt]);
    }
  }

  const int m = m0 + c;
  if (mode == 0) {
    __bf16* O = (__bf16*)outp;
    const int bb = m >> 10, s = m & 1023;
#pragma unroll
    for (int nt = 0; nt < 4; ++nt) {
      const int nbase = n0 + nt * 16 + hi * 8;
      const int h = nbase >> 6, d0 = nbase & 63;
      PK8 p;
#pragma unroll
      for (int i = 0; i < 8; ++i) p.e[i] = (__bf16)(acc[nt][i] + bias[nbase + i]);
      *(uint4*)(O + (((size_t)(bb * 8 + h) * 1024 + s) * 64 + d0)) = p.q;
    }
  } else {
    float* O = (float*)outp;
#pragma unroll
    for (int nt = 0; nt < 4; ++nt) {
      const int nbase = n0 + nt * 16 + hi * 8;
      float4 lo, hi4;
      lo.x  = acc[nt][0] + bias[nbase + 0];
      lo.y  = acc[nt][1] + bias[nbase + 1];
      lo.z  = acc[nt][2] + bias[nbase + 2];
      lo.w  = acc[nt][3] + bias[nbase + 3];
      hi4.x = acc[nt][4] + bias[nbase + 4];
      hi4.y = acc[nt][5] + bias[nbase + 5];
      hi4.z = acc[nt][6] + bias[nbase + 6];
      hi4.w = acc[nt][7] + bias[nbase + 7];
      *(float4*)(O + (size_t)m * 512 + nbase)     = lo;
      *(float4*)(O + (size_t)m * 512 + nbase + 4) = hi4;
    }
  }
}

// ---------------------------------------------------------------------------
// GEMM, normal orientation, output V^T [B,H,64,S].
// ---------------------------------------------------------------------------
__global__ void k_gemmN(const __bf16* __restrict__ X, const __bf16* __restrict__ Wt,
                        const float* __restrict__ bias, __bf16* __restrict__ Vt) {
  const int wave = threadIdx.x >> 5;
  const int lane = threadIdx.x & 31;
  const int c    = lane & 15;
  const int hi   = (lane >> 4) & 1;
  const int task = blockIdx.x * 8 + wave;
  const int m0   = (task >> 3) << 4;
  const int n0   = (task & 7)  << 6;

  v8f acc[4] = {};
  const __bf16* arow = X + (size_t)(m0 + c) * 512 + hi * 16;
  for (int ks = 0; ks < 512; ks += 32) {
    BV16 a;
    a.q[0] = *(const uint4*)(arow + ks);
    a.q[1] = *(const uint4*)(arow + ks + 8);
#pragma unroll
    for (int nt = 0; nt < 4; ++nt) {
      BV16 b;
      b.v = *(const v16bf*)(Wt + (size_t)(n0 + nt * 16 + c) * 512 + ks + hi * 16);
      acc[nt] = WMMA_BF16(a.v, b.v, acc[nt]);
    }
  }

  const int mrow = m0 + hi * 8;
  const int bb = mrow >> 10, s0 = mrow & 1023;
#pragma unroll
  for (int nt = 0; nt < 4; ++nt) {
    const int n = n0 + nt * 16 + c;
    const int h = n >> 6, d = n & 63;
    const float bn = bias[n];
    PK8 p;
#pragma unroll
    for (int i = 0; i < 8; ++i) p.e[i] = (__bf16)(acc[nt][i] + bn);
    *(uint4*)(Vt + (((size_t)(bb * 8 + h) * 64 + d) * 1024 + s0)) = p.q;
  }
}

// ---------------------------------------------------------------------------
// Flash attention per (b,h), TDM-staged K/V in LDS, double-buffered.
// LDS rows padded (K: 72 elems, V: 40 elems) for bank-conflict-free ds reads.
// ---------------------------------------------------------------------------
#define KROW 72                 // 64 data + 8 pad (16B pad per 128B row)
#define VROW 40                 // 32 data + 8 pad (16B pad per 64B row)
#define KBUF_E (32 * KROW)      // 2304 elems
#define BUF_E  (KBUF_E + 64 * VROW)  // 4864 elems = 9728 B per buffer

__global__ void k_attn(const __bf16* __restrict__ Q, const __bf16* __restrict__ K,
                       const __bf16* __restrict__ Vt, __bf16* __restrict__ M) {
  __shared__ __align__(16) __bf16 smem[2 * BUF_E];   // 19456 B

  const int wave = threadIdx.x >> 5;
  const int lane = threadIdx.x & 31;
  const int c    = lane & 15;
  const int hi   = (lane >> 4) & 1;
  const int hi8  = hi * 8;
  const int bh   = blockIdx.x >> 3;
  const int q0   = (blockIdx.x & 7) * 128 + wave * 16;

  const __bf16* Qh = Q  + (size_t)bh * 1024 * 64;
  const __bf16* Kh = K  + (size_t)bh * 1024 * 64;
  const __bf16* Vh = Vt + (size_t)bh * 64 * 1024;
  const unsigned lds_base = (unsigned)(size_t)(&smem[0]);   // low 32 bits = LDS offset

  // Q as B-operand (lane = query column, table-order d), kept in registers.
  BV16 qb0, qb1;
  {
    const __bf16* qp = Qh + (size_t)(q0 + c) * 64 + hi * 8;
    qb0.q[0] = *(const uint4*)(qp);      qb0.q[1] = *(const uint4*)(qp + 16);
    qb1.q[0] = *(const uint4*)(qp + 32); qb1.q[1] = *(const uint4*)(qp + 48);
  }

  // Prologue: stage tile 0. Wave0 -> K tile, wave1 -> V tile.
  if (wave == 0) {
    tdm_load_2d_bf16(lds_base, Kh, 64, 1024, 64, 64, 32, 4, 3, true);
  } else if (wave == 1) {
    tdm_load_2d_bf16(lds_base + KBUF_E * 2, Vh, 1024, 64, 1024, 32, 64, 3, 3, true);
  }

  v8f o[4] = {};
  float mrun = -1e30f, lrun = 0.f;

  for (int kt = 0; kt < 32; ++kt) {
    const int buf = kt & 1;
    const int t0  = kt * 32;
    if (wave == 0) {
      if (kt + 1 < 32) {
        tdm_load_2d_bf16(lds_base + (buf ^ 1) * BUF_E * 2,
                         Kh + (size_t)(t0 + 32) * 64, 64, 1024, 64, 64, 32, 4, 3, true);
        __builtin_amdgcn_s_wait_tensorcnt(1);
      } else {
        __builtin_amdgcn_s_wait_tensorcnt(0);
      }
    } else if (wave == 1) {
      if (kt + 1 < 32) {
        tdm_load_2d_bf16(lds_base + (buf ^ 1) * BUF_E * 2 + KBUF_E * 2,
                         Vh + (t0 + 32), 1024, 64, 1024, 32, 64, 3, 3, true);
        __builtin_amdgcn_s_wait_tensorcnt(1);
      } else {
        __builtin_amdgcn_s_wait_tensorcnt(0);
      }
    }
    __syncthreads();   // staged K/V visible to all waves

    const __bf16* kb = smem + buf * BUF_E;
    const __bf16* vb = kb + KBUF_E;

    // K as A-operand from LDS: 2 key tiles x 2 d-windows.
    const __bf16* kp0 = kb + c * KROW + hi8;
    const __bf16* kp1 = kb + (16 + c) * KROW + hi8;
    BV16 a00, a01, a10, a11;
    a00.q[0] = *(const uint4*)(kp0);      a00.q[1] = *(const uint4*)(kp0 + 16);
    a01.q[0] = *(const uint4*)(kp0 + 32); a01.q[1] = *(const uint4*)(kp0 + 48);
    a10.q[0] = *(const uint4*)(kp1);      a10.q[1] = *(const uint4*)(kp1 + 16);
    a11.q[0] = *(const uint4*)(kp1 + 32); a11.q[1] = *(const uint4*)(kp1 + 48);

    v8f s0 = {}, s1 = {};
    s0 = WMMA_BF16(a00.v, qb0.v, s0);
    s0 = WMMA_BF16(a01.v, qb1.v, s0);
    s1 = WMMA_BF16(a10.v, qb0.v, s1);
    s1 = WMMA_BF16(a11.v, qb1.v, s1);

    // online softmax over the key dimension (rows of the transposed tile)
    float tm = -1e30f;
#pragma unroll
    for (int i = 0; i < 8; ++i) {
      s0[i] *= 0.125f;  s1[i] *= 0.125f;
      tm = fmaxf(tm, fmaxf(s0[i], s1[i]));
    }
    tm = fmaxf(tm, __shfl_xor(tm, 16, 32));
    const float mnew  = fmaxf(mrun, tm);
    const float alpha = __expf(mrun - mnew);
    float ts = 0.f;
    BV16 pa;
#pragma unroll
    for (int i = 0; i < 8; ++i) {
      float p0 = __expf(s0[i] - mnew);
      float p1 = __expf(s1[i] - mnew);
      ts += p0 + p1;
      pa.e[i]     = (__bf16)p0;
      pa.e[8 + i] = (__bf16)p1;
    }
    ts += __shfl_xor(ts, 16, 32);
    lrun = lrun * alpha + ts;
    mrun = mnew;

#pragma unroll
    for (int i = 0; i < 8; ++i) {
      const float ai = __shfl(alpha, hi8 + i, 32);
      o[0][i] *= ai; o[1][i] *= ai; o[2][i] *= ai; o[3][i] *= ai;
    }

    // P @ V from LDS (V^T rows -> table-order t per lane).
#pragma unroll
    for (int nt = 0; nt < 4; ++nt) {
      const __bf16* vrow = vb + (size_t)(nt * 16 + c) * VROW + hi * 8;
      BV16 vbv;
      vbv.q[0] = *(const uint4*)(vrow);
      vbv.q[1] = *(const uint4*)(vrow + 16);
      o[nt] = WMMA_BF16(pa.v, vbv.v, o[nt]);
    }
    __syncthreads();   // all waves done reading buf before it is overwritten
  }

  const float inv = 1.0f / lrun;
  const int bb = bh >> 3, h = bh & 7;
#pragma unroll
  for (int i = 0; i < 8; ++i) {
    const float ai = __shfl(inv, hi8 + i, 32);
    const int s = q0 + hi8 + i;
    __bf16* mp = M + ((size_t)(bb * 1024 + s)) * 512 + h * 64 + c;
    mp[0]  = (__bf16)(o[0][i] * ai);
    mp[16] = (__bf16)(o[1][i] * ai);
    mp[32] = (__bf16)(o[2][i] * ai);
    mp[48] = (__bf16)(o[3][i] * ai);
  }
}

// ---------------------------------------------------------------------------
extern "C" void kernel_launch(void* const* d_in, const int* in_sizes, int n_in,
                              void* d_out, int out_size, void* d_ws, size_t ws_size,
                              hipStream_t stream) {
  (void)in_sizes; (void)n_in; (void)out_size; (void)ws_size;
  const float* nodes = (const float*)d_in[0];
  const float* Wq = (const float*)d_in[1];
  const float* Wk = (const float*)d_in[2];
  const float* Wv = (const float*)d_in[3];
  const float* bq = (const float*)d_in[4];
  const float* bk = (const float*)d_in[5];
  const float* bv = (const float*)d_in[6];
  const float* Wo = (const float*)d_in[7];
  const float* bo = (const float*)d_in[8];
  float* out = (float*)d_out;

  char* ws = (char*)d_ws;
  __bf16* Xb  = (__bf16*)(ws + 0);                 //  8 MB  [8192,512]
  __bf16* Wqt = (__bf16*)(ws + (8u  << 20));
  __bf16* Wkt = (__bf16*)(ws + (8u  << 20) + (512u << 10));
  __bf16* Wvt = (__bf16*)(ws + (9u  << 20));
  __bf16* Wot = (__bf16*)(ws + (9u  << 20) + (512u << 10));
  __bf16* Qb  = (__bf16*)(ws + (10u << 20));       //  8 MB  [B,H,S,64]
  __bf16* Kb  = (__bf16*)(ws + (18u << 20));       //  8 MB  [B,H,S,64]
  __bf16* Vtb = (__bf16*)(ws + (26u << 20));       //  8 MB  [B,H,64,S]
  __bf16* Mb  = (__bf16*)(ws + (34u << 20));       //  8 MB  [8192,512]

  k_cvt<<<4096, 256, 0, stream>>>(nodes, Xb, 8192 * 512);
  k_tw <<<1024, 256, 0, stream>>>(Wq, Wqt);
  k_tw <<<1024, 256, 0, stream>>>(Wk, Wkt);
  k_tw <<<1024, 256, 0, stream>>>(Wv, Wvt);
  k_tw <<<1024, 256, 0, stream>>>(Wo, Wot);

  k_gemmT<<<512, 256, 0, stream>>>(Wqt, Xb, bq, (void*)Qb, 0);
  k_gemmT<<<512, 256, 0, stream>>>(Wkt, Xb, bk, (void*)Kb, 0);
  k_gemmN<<<512, 256, 0, stream>>>(Xb, Wvt, bv, Vtb);

  k_attn <<<512, 256, 0, stream>>>(Qb, Kb, Vtb, Mb);

  k_gemmT<<<512, 256, 0, stream>>>(Wot, Mb, bo, (void*)out, 1);
}
